// TamiyoGNN_6743098655550
// MI455X (gfx1250) — matrix operand order repaired and checked by reading, not compiled, not run on hardware
//
#include <hip/hip_runtime.h>
#include <hip/hip_bf16.h>
#include <math.h>
#include <stdint.h>

// ---------------- problem constants ----------------
#define E_EDGES 100000
#define HS 256      // SAGE hidden dim
#define HG 128      // GAT hidden dim
#define NHEAD 4
#define NTYPES 6
#define NRELS 21
#define NTOT 150512 // sum of node counts

typedef __attribute__((ext_vector_type(16))) __bf16 v16bf;
typedef __attribute__((ext_vector_type(8)))  __bf16 v8bf;
typedef __attribute__((ext_vector_type(8)))  float  v8f;
typedef __attribute__((ext_vector_type(4)))  unsigned v4u;
typedef __attribute__((ext_vector_type(8)))  int    v8i;
typedef __attribute__((ext_vector_type(4)))  int    v4i;

// ---------------- device helpers ----------------
__device__ __forceinline__ float gelu_f(float x) {
  return 0.5f * x * (1.0f + erff(x * 0.70710678118654752f));
}
// monotonic float<->uint encoding for atomic max over signed floats
__device__ __forceinline__ unsigned enc_f(float x) {
  unsigned u = __float_as_uint(x);
  return (u & 0x80000000u) ? ~u : (u | 0x80000000u);
}
__device__ __forceinline__ float dec_f(unsigned k) {
  return (k & 0x80000000u) ? __uint_as_float(k & 0x7fffffffu)
                           : __uint_as_float(~k);
}

// ---------------- Tensor Data Mover: 2D bf16 tile, global -> LDS ----------------
// D# per CDNA5 ISA ch.8: group0 = {count/lds_addr/global_addr/type},
// group1 = {data_size, pad, tensor dims (OOB -> zero), tile dims, stride}.
// pad_interval=16 dwords (one 32-elem bf16 row), pad_amount=4 dwords (16B)
// => LDS row stride 80B, matching APAD=40 below.
__device__ __forceinline__ void tdm_load_2d_bf16(
    unsigned lds_off, const __bf16* gptr,
    unsigned tileK, unsigned tileRows,
    unsigned tdimK, unsigned tdimRows,
    unsigned long strideElems)
{
  unsigned long ga = (unsigned long)(size_t)(const void*)gptr;
  v4u g0;
  g0[0] = 1u;                                    // count=1 (valid user descriptor)
  g0[1] = lds_off;                               // lds_addr [63:32]
  g0[2] = (unsigned)ga;                          // global_addr [95:64]
  g0[3] = (unsigned)((ga >> 32) & 0x1ffffffu) | (2u << 30);  // addr[56:32] | type=2
  v8i g1;
  g1[0] = (int)((1u << 16) | (1u << 20) | (3u << 22) | (3u << 25));
      // data_size=1 (2B) | pad_enable | pad_interval=3 (16 dw) | pad_amount=3 (4 dw)
  g1[1] = (int)((tdimK & 0xffffu) << 16);                          // tensor_dim0[15:0]
  g1[2] = (int)(((tdimK >> 16) & 0xffffu) | ((tdimRows & 0xffffu) << 16));
  g1[3] = (int)(((tdimRows >> 16) & 0xffffu) | ((tileK & 0xffffu) << 16)); // tile_dim0
  g1[4] = (int)(tileRows & 0xffffu);                               // tile_dim1 (dim2=0)
  g1[5] = (int)(strideElems & 0xffffffffu);                        // dim0_stride lo
  g1[6] = (int)((strideElems >> 32) & 0xffffu);                    // dim0_stride hi
  g1[7] = 0;                                                       // dim1_stride=0 (2D)
  v4i gz = {0, 0, 0, 0};
#if defined(__clang_major__) && (__clang_major__ >= 23)
  v8i gz8 = {0, 0, 0, 0, 0, 0, 0, 0};
  __builtin_amdgcn_tensor_load_to_lds(g0, g1, gz, gz, gz8, 0);
#else
  __builtin_amdgcn_tensor_load_to_lds(g0, g1, gz, gz, 0);
#endif
}

// ---------------- WMMA GEMM: C = [C +] A(MxK) * Bt(NxK)^T [+ bias] ----------------
// A: bf16 row-major MxK. Bt: bf16 N-major (pre-transposed weights) NxK.
// Double-buffered TDM staging: the k+1 tile pair is DMA'd while tile k is
// consumed by WMMAs (TENSORcnt retires in order, so wait<=2 == "current tile
// landed, next still in flight"). Fragments are pure ds_load_b128;
// 4 v_wmma per wave per K-step.
// Assumptions (all launches satisfy them): K % 32 == 0, N % 64 == 0.
#define BM 128
#define BN 64
#define BK 32
#define APAD 40   // LDS row stride in bf16 elems (80B)

__device__ __forceinline__ v16bf load_a_frag(const __bf16* As, int rowBase, int lane) {
  // 16-bit A 16x32 layout (ISA 7.12.2): lane m=lane&15, hf=lane>>4;
  // VGPR v<4 -> K = hf*8 + 2v ; v>=4 -> K = 16 + hf*8 + 2(v-4)
  int m = rowBase + (lane & 15);
  int hf = lane >> 4;
  v8bf a0 = *(const v8bf*)(As + m * APAD + hf * 8);
  v8bf a1 = *(const v8bf*)(As + m * APAD + 16 + hf * 8);
  return __builtin_shufflevector(a0, a1, 0, 1, 2, 3, 4, 5, 6, 7,
                                 8, 9, 10, 11, 12, 13, 14, 15);
}

__device__ __forceinline__ v16bf load_b_frag(const __bf16* Bs, int colBase, int lane) {
  // B 32x16 layout (matches A): lane n=lane&15, hf=lane>>4;
  // elems 0-7 -> K = hf*8 + s ; elems 8-15 -> K = 16 + hf*8 + (s-8).
  int n = colBase + (lane & 15);
  int kb = (lane >> 4) * 8;
  v8bf b0 = *(const v8bf*)(Bs + n * APAD + kb);
  v8bf b1 = *(const v8bf*)(Bs + n * APAD + 16 + kb);
  return __builtin_shufflevector(b0, b1, 0, 1, 2, 3, 4, 5, 6, 7,
                                 8, 9, 10, 11, 12, 13, 14, 15);
}

__device__ __forceinline__ void store_frag(const v8f& acc, float* __restrict__ C,
                                           int ldc, int M, int row0, int col0,
                                           const float* __restrict__ bias,
                                           int accumulate, int lane) {
  // C/D layout: lane n = lane&15; VGPR r -> row = r + (lane>>4)*8
  int n = lane & 15, hf = lane >> 4;
  #pragma unroll
  for (int r = 0; r < 8; ++r) {
    int row = row0 + hf * 8 + r;
    int col = col0 + n;             // always < N (N % 64 == 0)
    if (row < M) {
      float v = acc[r];
      if (bias) v += bias[col];
      size_t idx = (size_t)row * ldc + col;
      C[idx] = accumulate ? (C[idx] + v) : v;
    }
  }
}

__global__ __launch_bounds__(256) void gemm_wmma_kernel(
    const __bf16* __restrict__ A, int lda,
    const __bf16* __restrict__ Btg, int ldbt,
    float* __restrict__ C, int ldc,
    int M, int N, int K,
    const float* __restrict__ bias, int accumulate)
{
  __shared__ __bf16 As[2][BM * APAD];   // 2 x 10240 B
  __shared__ __bf16 Bs[2][BN * APAD];   // 2 x  5120 B
  const int tid  = threadIdx.x;
  const int lane = tid & 31;
  const int wid  = tid >> 5;
  const int waveRow = (wid & 3) * 32;   // 4 wave-rows of 32
  const int waveCol = (wid >> 2) * 32;  // 2 wave-cols of 32
  const int blockRow = blockIdx.x * BM;
  const int blockCol = blockIdx.y * BN;
  unsigned asOff[2] = {(unsigned)(size_t)(void*)&As[0][0],
                       (unsigned)(size_t)(void*)&As[1][0]};
  unsigned bsOff[2] = {(unsigned)(size_t)(void*)&Bs[0][0],
                       (unsigned)(size_t)(void*)&Bs[1][0]};

  v8f acc00 = {}, acc01 = {}, acc10 = {}, acc11 = {};

  const int nk = K / BK;
  // prologue: DMA tile 0 into buffer 0
  if (wid == 0) {
    tdm_load_2d_bf16(asOff[0], A + (size_t)blockRow * lda,
                     BK, BM, (unsigned)K, (unsigned)(M - blockRow),
                     (unsigned long)lda);
    tdm_load_2d_bf16(bsOff[0], Btg + (size_t)blockCol * ldbt,
                     BK, BN, (unsigned)K, (unsigned)(N - blockCol),
                     (unsigned long)ldbt);
  }

  for (int ki = 0; ki < nk; ++ki) {
    const int cur = ki & 1;
    if (wid == 0) {
      if (ki + 1 < nk) {
        int k1 = (ki + 1) * BK;
        tdm_load_2d_bf16(asOff[cur ^ 1], A + (size_t)blockRow * lda + k1,
                         BK, BM, (unsigned)(K - k1), (unsigned)(M - blockRow),
                         (unsigned long)lda);
        tdm_load_2d_bf16(bsOff[cur ^ 1], Btg + (size_t)blockCol * ldbt + k1,
                         BK, BN, (unsigned)(K - k1), (unsigned)(N - blockCol),
                         (unsigned long)ldbt);
        __builtin_amdgcn_s_wait_tensorcnt(2);  // tile ki landed; ki+1 in flight
      } else {
        __builtin_amdgcn_s_wait_tensorcnt(0);
      }
    }
    __syncthreads();

    v16bf a0 = load_a_frag(&As[cur][0], waveRow, lane);
    v16bf a1 = load_a_frag(&As[cur][0], waveRow + 16, lane);
    v16bf b0 = load_b_frag(&Bs[cur][0], waveCol, lane);
    v16bf b1 = load_b_frag(&Bs[cur][0], waveCol + 16, lane);

    acc00 = __builtin_amdgcn_wmma_f32_16x16x32_bf16(false, a0, false, b0, (short)0, acc00, false, false);
    acc01 = __builtin_amdgcn_wmma_f32_16x16x32_bf16(false, a0, false, b1, (short)0, acc01, false, false);
    acc10 = __builtin_amdgcn_wmma_f32_16x16x32_bf16(false, a1, false, b0, (short)0, acc10, false, false);
    acc11 = __builtin_amdgcn_wmma_f32_16x16x32_bf16(false, a1, false, b1, (short)0, acc11, false, false);
    __syncthreads();   // all waves done reading buffer `cur` before it is refilled
  }

  store_frag(acc00, C, ldc, M, blockRow + waveRow,      blockCol + waveCol,      bias, accumulate, lane);
  store_frag(acc01, C, ldc, M, blockRow + waveRow,      blockCol + waveCol + 16, bias, accumulate, lane);
  store_frag(acc10, C, ldc, M, blockRow + waveRow + 16, blockCol + waveCol,      bias, accumulate, lane);
  store_frag(acc11, C, ldc, M, blockRow + waveRow + 16, blockCol + waveCol + 16, bias, accumulate, lane);
}

static inline void launch_gemm(const __bf16* A, int lda, const __bf16* Bt, int ldbt,
                               float* C, int ldc, int M, int N, int K,
                               const float* bias, int acc, hipStream_t s) {
  dim3 grid((M + BM - 1) / BM, (N + BN - 1) / BN);
  gemm_wmma_kernel<<<grid, 256, 0, s>>>(A, lda, Bt, ldbt, C, ldc, M, N, K, bias, acc);
}

// ---------------- weight convert + transpose: W(KxN) f32 -> Wt(NxK) bf16 ----------
__global__ void convt_kernel(const float* __restrict__ W, __bf16* __restrict__ Wt,
                             int K, int N) {
  long i = (long)blockIdx.x * 256 + threadIdx.x;
  if (i >= (long)K * N) return;
  int k = (int)(i / N), n = (int)(i % N);
  Wt[(size_t)n * K + k] = (__bf16)W[i];
}

// ---------------- encoder: h = LN(gelu(x@W + b)), with bf16 mirror ----------------
__global__ __launch_bounds__(256) void encoder_kernel(
    const float* __restrict__ x, const float* __restrict__ W,
    const float* __restrict__ b, const float* __restrict__ gamma,
    const float* __restrict__ beta, float* __restrict__ h,
    __bf16* __restrict__ hb, int inDim)
{
  int node = blockIdx.x;
  int d = threadIdx.x;
  float acc = b[d];
  for (int k = 0; k < inDim; ++k)
    acc += x[(size_t)node * inDim + k] * W[k * HS + d];
  float v = gelu_f(acc);
  __shared__ float sm[HS];
  __shared__ float stat[2];
  sm[d] = v; __syncthreads();
  for (int sft = 128; sft > 0; sft >>= 1) { if (d < sft) sm[d] += sm[d + sft]; __syncthreads(); }
  if (d == 0) stat[0] = sm[0] * (1.0f / HS);
  __syncthreads();
  float mu = stat[0];
  float dv = v - mu;
  sm[d] = dv * dv; __syncthreads();
  for (int sft = 128; sft > 0; sft >>= 1) { if (d < sft) sm[d] += sm[d + sft]; __syncthreads(); }
  if (d == 0) stat[1] = sm[0] * (1.0f / HS);
  __syncthreads();
  float o = dv * rsqrtf(stat[1] + 1e-5f) * gamma[d] + beta[d];
  h[(size_t)node * HS + d] = o;
  hb[(size_t)node * HS + d] = (__bf16)o;
}

// ---------------- SAGE ----------------
__global__ __launch_bounds__(256) void sage_scatter_kernel(
    const int* __restrict__ si, const int* __restrict__ di,
    const float* __restrict__ hs, float* __restrict__ agg, float* __restrict__ deg)
{
  int e = blockIdx.x; int d = threadIdx.x;
  int s = si[e], t = di[e];
  atomicAdd(&agg[(size_t)t * HS + d], hs[(size_t)s * HS + d]);
  if (d == 0) atomicAdd(&deg[t], 1.0f);
}

// normalize by degree and emit bf16 (GEMM operand)
__global__ void sage_norm_kernel(const float* __restrict__ agg,
                                 const float* __restrict__ deg,
                                 __bf16* __restrict__ aggb, long n) {
  long i = (long)blockIdx.x * 256 + threadIdx.x;
  if (i >= n) return;
  aggb[i] = (__bf16)(agg[i] / fmaxf(deg[i >> 8], 1.0f));   // HS == 256
}

__global__ void sage_finalize_kernel(float* __restrict__ h, __bf16* __restrict__ hb,
                                     const float* __restrict__ out,
                                     float invcnt, long n) {
  long i = (long)blockIdx.x * 256 + threadIdx.x;
  if (i >= n) return;
  float v = gelu_f(out[i] * invcnt) + h[i];
  h[i] = v;
  hb[i] = (__bf16)v;
}

// ---------------- GAT ----------------
// Fold attention vectors into weights: ws_vec[k][h] = sum_d Wsrc[k, h*HG+d]*a_src[h,d]
__global__ __launch_bounds__(256) void gat_prep_kernel(
    const float* __restrict__ Wsrc, const float* __restrict__ Wdst,
    const float* __restrict__ We,
    const float* __restrict__ a_s, const float* __restrict__ a_d,
    const float* __restrict__ a_e,
    float* __restrict__ ws_vec, float* __restrict__ wd_vec,
    float* __restrict__ t_edge, int ind)
{
  int tid = threadIdx.x;
  for (int i = tid; i < ind * NHEAD; i += 256) {
    int k = i >> 2, h = i & 3;
    float s1 = 0.f, s2 = 0.f;
    for (int d = 0; d < HG; ++d) {
      s1 += Wsrc[(size_t)k * (NHEAD * HG) + h * HG + d] * a_s[h * HG + d];
      s2 += Wdst[(size_t)k * (NHEAD * HG) + h * HG + d] * a_d[h * HG + d];
    }
    ws_vec[i] = s1; wd_vec[i] = s2;
  }
  if (tid < 12) {  // t_edge[c*4+h], c<3
    int c = tid >> 2, h = tid & 3;
    float s = 0.f;
    for (int d = 0; d < HG; ++d)
      s += We[(size_t)c * (NHEAD * HG) + h * HG + d] * a_e[h * HG + d];
    t_edge[tid] = s;
  }
}

// al[n][h] = sum_k h[n,k] * vec[k*4+h]
__global__ void gemv4_kernel(const float* __restrict__ h, int dim,
                             const float* __restrict__ vec,
                             float* __restrict__ al, int n)
{
  int i = blockIdx.x * 256 + threadIdx.x;
  if (i >= n * NHEAD) return;
  int node = i >> 2, hh = i & 3;
  float s = 0.f;
  for (int k = 0; k < dim; ++k) s += h[(size_t)node * dim + k] * vec[k * 4 + hh];
  al[i] = s;
}

__global__ void gat_logit_kernel(const int* __restrict__ si, const int* __restrict__ di,
                                 const float* __restrict__ ea,
                                 const float* __restrict__ t_edge,
                                 const float* __restrict__ al_s,
                                 const float* __restrict__ al_d,
                                 float* __restrict__ logit, unsigned* __restrict__ menc)
{
  int i = blockIdx.x * 256 + threadIdx.x;
  if (i >= E_EDGES * NHEAD) return;
  int e = i >> 2, hh = i & 3;
  float le = ea[(size_t)e * 3 + 0] * t_edge[0 * 4 + hh]
           + ea[(size_t)e * 3 + 1] * t_edge[1 * 4 + hh]
           + ea[(size_t)e * 3 + 2] * t_edge[2 * 4 + hh];
  float x = al_s[si[e] * 4 + hh] + al_d[di[e] * 4 + hh] + le;
  x = (x > 0.f) ? x : 0.2f * x;  // leaky_relu(0.2)
  logit[i] = x;
  atomicMax(&menc[di[e] * 4 + hh], enc_f(x));
}

__global__ void gat_w_kernel(const int* __restrict__ di,
                             const float* __restrict__ logit,
                             const unsigned* __restrict__ menc,
                             float* __restrict__ w, float* __restrict__ den)
{
  int i = blockIdx.x * 256 + threadIdx.x;
  if (i >= E_EDGES * NHEAD) return;
  int e = i >> 2, hh = i & 3;
  float m = dec_f(menc[di[e] * 4 + hh]);
  float ww = __expf(logit[i] - m);
  w[i] = ww;
  atomicAdd(&den[di[e] * 4 + hh], ww);
}

__global__ __launch_bounds__(128) void gat_scatter_kernel(
    const int* __restrict__ si, const int* __restrict__ di,
    const float* __restrict__ xs, const float* __restrict__ w,
    const float* __restrict__ den, float* __restrict__ out)
{
  int e = blockIdx.x; int d = threadIdx.x;
  int s = si[e], t = di[e];
  float acc = 0.f;
  #pragma unroll
  for (int hh = 0; hh < NHEAD; ++hh) {
    float alpha = w[e * 4 + hh] / den[t * 4 + hh];
    acc += alpha * xs[(size_t)s * (NHEAD * HG) + hh * HG + d];
  }
  atomicAdd(&out[(size_t)t * HG + d], 0.25f * acc);  // mean over heads
}

__global__ void gat_bias_kernel(float* out, const float* __restrict__ bias, long n) {
  long i = (long)blockIdx.x * 256 + threadIdx.x;
  if (i >= n) return;
  out[i] += bias[i & (HG - 1)];
}

__global__ void gat_finalize_kernel(float* __restrict__ h, __bf16* __restrict__ hb,
                                    const float* __restrict__ out,
                                    float invcnt, int hasRes, long n) {
  long i = (long)blockIdx.x * 256 + threadIdx.x;
  if (i >= n) return;
  float v = gelu_f(out[i] * invcnt);
  if (hasRes) v += h[i];
  h[i] = v;
  hb[i] = (__bf16)v;
}

// ---------------- heads ----------------
struct HeadsArgs {
  const float* hg;
  const float *policy_W, *policy_b, *param_W1, *param_b1, *param_W2, *param_b2;
  const float *blend_W, *blend_b, *sched_W1, *sched_b1, *sched_W2, *sched_b2;
  const float *breaker_W, *breaker_b, *risk_W, *risk_b, *value_W, *value_b;
  float* out;
};

__global__ __launch_bounds__(256) void heads_kernel(HeadsArgs a) {
  __shared__ float g[HG];
  __shared__ float t1[64];
  __shared__ float t2[64];
  int tid = threadIdx.x;
  if (tid < HG) {
    float s = 0.f;
    for (int n = 0; n < 512; ++n) s += a.hg[(size_t)n * HG + tid];
    g[tid] = s * (1.0f / 512.0f);
    a.out[47 + tid] = g[tid];
  }
  __syncthreads();
  if (tid < 32) {
    float s = a.policy_b[tid];
    for (int k = 0; k < HG; ++k) s += g[k] * a.policy_W[k * 32 + tid];
    a.out[tid] = s;
  } else if (tid < 96) {
    int j = tid - 32;
    float s = a.param_b1[j];
    for (int k = 0; k < HG; ++k) s += g[k] * a.param_W1[k * 64 + j];
    t1[j] = gelu_f(s);
  } else if (tid < 160) {
    int j = tid - 96;
    float s = a.sched_b1[j];
    for (int k = 0; k < HG; ++k) s += g[k] * a.sched_W1[k * 64 + j];
    t2[j] = gelu_f(s);
  } else if (tid < 163) {
    int j = tid - 160;
    float s = a.blend_b[j];
    for (int k = 0; k < HG; ++k) s += g[k] * a.blend_W[k * 3 + j];
    a.out[36 + j] = s;
  } else if (tid < 168) {
    int j = tid - 163;
    float s = a.risk_b[j];
    for (int k = 0; k < HG; ++k) s += g[k] * a.risk_W[k * 5 + j];
    a.out[39 + j] = s;
  } else if (tid == 168) {
    float s = a.value_b[0];
    for (int k = 0; k < HG; ++k) s += g[k] * a.value_W[k];
    a.out[44] = s;
  } else if (tid < 171) {
    int j = tid - 169;
    float s = a.breaker_b[j];
    for (int k = 0; k < HG; ++k) s += g[k] * a.breaker_W[k * 2 + j];
    a.out[30176 + j] = s;
  }
  __syncthreads();
  if (tid < 4) {
    float s = a.param_b2[tid];
    for (int k = 0; k < 64; ++k) s += t1[k] * a.param_W2[k * 4 + tid];
    a.out[32 + tid] = tanhf(s);
  } else if (tid < 6) {
    int j = tid - 4;
    float s = a.sched_b2[j];
    for (int k = 0; k < 64; ++k) s += t2[k] * a.sched_W2[k * 2 + j];
    a.out[45 + j] = tanhf(s);
  }
  __syncthreads();
  if (tid == 0) a.out[175] = a.out[32];  // param_delta = policy_params[...,0]
}

__global__ void score_kernel(const float* __restrict__ h, const float* __restrict__ W,
                             const float* __restrict__ b, float* __restrict__ out, int n) {
  int i = blockIdx.x * 256 + threadIdx.x;
  if (i >= n) return;
  float s = b[0];
  for (int k = 0; k < HG; ++k) s += h[(size_t)i * HG + k] * W[k];
  out[i] = s;
}

// ---------------- host orchestration ----------------
extern "C" void kernel_launch(void* const* d_in, const int* in_sizes, int n_in,
                              void* d_out, int out_size, void* d_ws, size_t ws_size,
                              hipStream_t stream) {
  (void)in_sizes; (void)n_in; (void)out_size; (void)ws_size;

  static const int    Ns[NTYPES]      = {512, 20000, 10000, 50000, 40000, 30000};
  static const int    IND[NTYPES]     = {16, 14, 14, 12, 8, 10};
  static const size_t nodeOff[NTYPES] = {0, 512, 20512, 30512, 80512, 120512};
  static const int    RELS[NRELS][2]  = {
    {0,1},{1,0},{0,2},{2,0},{1,1},{0,3},{3,0},{3,4},{4,3},{4,5},{5,4},
    {1,5},{5,1},{2,3},{3,2},{2,4},{5,2},{3,3},{1,3},{3,1},{3,4}};
  static const float  INVCNT[NTYPES]  = {1.f/3.f, 1.f/4.f, 1.f/3.f, 1.f/5.f, 1.f/4.f, 1.f/2.f};

  const float* x[NTYPES];
  for (int t = 0; t < NTYPES; ++t) x[t] = (const float*)d_in[t];
  const int*   ei = (const int*)d_in[6];     // (21, 2, E)
  const float* ea = (const float*)d_in[7];   // (21, E, 3)
  const float* sage_Wl = (const float*)d_in[32];
  const float* sage_bl = (const float*)d_in[33];
  const float* sage_Wr = (const float*)d_in[34];
  float* out = (float*)d_out;

  // workspace layout (~525 MB)
  char* ws = (char*)d_ws;
  size_t off = 0;
  auto take = [&](size_t bytes) { size_t o = off; off += (bytes + 255) & ~(size_t)255; return o; };
  float*    buf_h     = (float*)   (ws + take((size_t)NTOT * HS * 4));
  float*    buf_out   = (float*)   (ws + take((size_t)NTOT * HS * 4));
  float*    buf_xs    = (float*)   (ws + take((size_t)50000 * 512 * 4));  // also SAGE agg
  __bf16*   buf_hb    = (__bf16*)  (ws + take((size_t)NTOT * HS * 2));    // bf16 mirror of h
  __bf16*   buf_aggb  = (__bf16*)  (ws + take((size_t)50000 * HS * 2));   // bf16 norm. agg
  __bf16*   buf_wt1   = (__bf16*)  (ws + take((size_t)512 * 256 * 2));    // bf16 Wt
  __bf16*   buf_wt2   = (__bf16*)  (ws + take((size_t)512 * 256 * 2));
  float*    buf_deg   = (float*)   (ws + take((size_t)50000 * 4));
  float*    buf_als   = (float*)   (ws + take((size_t)50000 * 16));
  float*    buf_ald   = (float*)   (ws + take((size_t)50000 * 16));
  unsigned* buf_m     = (unsigned*)(ws + take((size_t)50000 * 16));
  float*    buf_den   = (float*)   (ws + take((size_t)50000 * 16));
  float*    buf_logit = (float*)   (ws + take((size_t)E_EDGES * 16));
  float*    buf_w     = (float*)   (ws + take((size_t)E_EDGES * 16));
  float*    buf_wsvec = (float*)   (ws + take(256 * 4 * 4));
  float*    buf_wdvec = (float*)   (ws + take(256 * 4 * 4));
  float*    buf_tedge = (float*)   (ws + take(64));

  // ---- encoders ----
  for (int t = 0; t < NTYPES; ++t) {
    const float* eW  = (const float*)d_in[8 + 4 * t + 0];
    const float* eb  = (const float*)d_in[8 + 4 * t + 1];
    const float* eg  = (const float*)d_in[8 + 4 * t + 2];
    const float* ebe = (const float*)d_in[8 + 4 * t + 3];
    encoder_kernel<<<Ns[t], 256, 0, stream>>>(x[t], eW, eb, eg, ebe,
                                              buf_h + nodeOff[t] * HS,
                                              buf_hb + nodeOff[t] * HS, IND[t]);
  }

  // ---- 2 SAGE layers ----
  for (int l = 0; l < 2; ++l) {
    hipMemsetAsync(buf_out, 0, (size_t)NTOT * HS * 4, stream);
    for (int r = 0; r < NRELS; ++r) {
      int s = RELS[r][0], t = RELS[r][1];
      const int* si = ei + (size_t)r * 2 * E_EDGES;
      const int* di = si + E_EDGES;
      hipMemsetAsync(buf_xs, 0, (size_t)Ns[t] * HS * 4, stream);
      hipMemsetAsync(buf_deg, 0, (size_t)Ns[t] * 4, stream);
      sage_scatter_kernel<<<E_EDGES, 256, 0, stream>>>(
          si, di, buf_h + nodeOff[s] * HS, buf_xs, buf_deg);
      long nelt = (long)Ns[t] * HS;
      sage_norm_kernel<<<(nelt + 255) / 256, 256, 0, stream>>>(
          buf_xs, buf_deg, buf_aggb, nelt);
      const float* Wl = sage_Wl + ((size_t)l * NRELS + r) * HS * HS;
      const float* bl = sage_bl + ((size_t)l * NRELS + r) * HS;
      const float* Wr = sage_Wr + ((size_t)l * NRELS + r) * HS * HS;
      convt_kernel<<<(HS * HS + 255) / 256, 256, 0, stream>>>(Wl, buf_wt1, HS, HS);
      convt_kernel<<<(HS * HS + 255) / 256, 256, 0, stream>>>(Wr, buf_wt2, HS, HS);
      launch_gemm(buf_aggb, HS, buf_wt1, HS, buf_out + nodeOff[t] * HS, HS,
                  Ns[t], HS, HS, bl, 1, stream);
      launch_gemm(buf_hb + nodeOff[t] * HS, HS, buf_wt2, HS,
                  buf_out + nodeOff[t] * HS, HS, Ns[t], HS, HS, nullptr, 1, stream);
    }
    for (int t = 0; t < NTYPES; ++t) {
      long nelt = (long)Ns[t] * HS;
      sage_finalize_kernel<<<(nelt + 255) / 256, 256, 0, stream>>>(
          buf_h + nodeOff[t] * HS, buf_hb + nodeOff[t] * HS,
          buf_out + nodeOff[t] * HS, INVCNT[t], nelt);
    }
  }

  // ---- 2 GAT layers ----
  for (int l = 0; l < 2; ++l) {
    int ind = (l == 0) ? HS : HG;
    const float* Wsrc = (const float*)d_in[35 + 7 * l + 0];
    const float* Wdst = (const float*)d_in[35 + 7 * l + 1];
    const float* We   = (const float*)d_in[35 + 7 * l + 2];
    const float* a_s  = (const float*)d_in[35 + 7 * l + 3];
    const float* a_d  = (const float*)d_in[35 + 7 * l + 4];
    const float* a_e  = (const float*)d_in[35 + 7 * l + 5];
    const float* gb   = (const float*)d_in[35 + 7 * l + 6];
    hipMemsetAsync(buf_out, 0, (size_t)NTOT * HG * 4, stream);
    for (int r = 0; r < NRELS; ++r) {
      int s = RELS[r][0], t = RELS[r][1];
      const int* si = ei + (size_t)r * 2 * E_EDGES;
      const int* di = si + E_EDGES;
      const float* Wsr = Wsrc + (size_t)r * ind * (NHEAD * HG);
      const float* Wdr = Wdst + (size_t)r * ind * (NHEAD * HG);
      const float* Wer = We + (size_t)r * 3 * (NHEAD * HG);
      const float* ear = ea + (size_t)r * E_EDGES * 3;
      gat_prep_kernel<<<1, 256, 0, stream>>>(Wsr, Wdr, Wer,
          a_s + (size_t)r * (NHEAD * HG), a_d + (size_t)r * (NHEAD * HG),
          a_e + (size_t)r * (NHEAD * HG), buf_wsvec, buf_wdvec, buf_tedge, ind);
      // xs = h[s] @ Wsrc  (M=N[s], K=ind, N=512) — WMMA via double-buffered TDM
      convt_kernel<<<(ind * NHEAD * HG + 255) / 256, 256, 0, stream>>>(
          Wsr, buf_wt1, ind, NHEAD * HG);
      launch_gemm(buf_hb + nodeOff[s] * ind, ind, buf_wt1, ind,
                  buf_xs, NHEAD * HG, Ns[s], NHEAD * HG, ind, nullptr, 0, stream);
      gemv4_kernel<<<(Ns[s] * 4 + 255) / 256, 256, 0, stream>>>(
          buf_h + nodeOff[s] * ind, ind, buf_wsvec, buf_als, Ns[s]);
      gemv4_kernel<<<(Ns[t] * 4 + 255) / 256, 256, 0, stream>>>(
          buf_h + nodeOff[t] * ind, ind, buf_wdvec, buf_ald, Ns[t]);
      hipMemsetAsync(buf_m, 0, (size_t)Ns[t] * 16, stream);     // 0 < enc(-inf)
      hipMemsetAsync(buf_den, 0, (size_t)Ns[t] * 16, stream);
      gat_logit_kernel<<<(E_EDGES * 4 + 255) / 256, 256, 0, stream>>>(
          si, di, ear, buf_tedge, buf_als, buf_ald, buf_logit, buf_m);
      gat_w_kernel<<<(E_EDGES * 4 + 255) / 256, 256, 0, stream>>>(
          di, buf_logit, buf_m, buf_w, buf_den);
      gat_scatter_kernel<<<E_EDGES, 128, 0, stream>>>(
          si, di, buf_xs, buf_w, buf_den, buf_out + nodeOff[t] * HG);
      long nelt = (long)Ns[t] * HG;
      gat_bias_kernel<<<(nelt + 255) / 256, 256, 0, stream>>>(
          buf_out + nodeOff[t] * HG, gb + (size_t)r * HG, nelt);
    }
    for (int t = 0; t < NTYPES; ++t) {
      long nelt = (long)Ns[t] * HG;
      gat_finalize_kernel<<<(nelt + 255) / 256, 256, 0, stream>>>(
          buf_h + nodeOff[t] * HG, buf_hb + nodeOff[t] * HG,
          buf_out + nodeOff[t] * HG, INVCNT[t], (l == 1) ? 1 : 0, nelt);
    }
  }

  // ---- heads ----
  HeadsArgs ha;
  ha.hg = buf_h + nodeOff[0] * HG;
  ha.policy_W  = (const float*)d_in[49]; ha.policy_b  = (const float*)d_in[50];
  ha.param_W1  = (const float*)d_in[51]; ha.param_b1  = (const float*)d_in[52];
  ha.param_W2  = (const float*)d_in[53]; ha.param_b2  = (const float*)d_in[54];
  ha.blend_W   = (const float*)d_in[55]; ha.blend_b   = (const float*)d_in[56];
  ha.sched_W1  = (const float*)d_in[57]; ha.sched_b1  = (const float*)d_in[58];
  ha.sched_W2  = (const float*)d_in[59]; ha.sched_b2  = (const float*)d_in[60];
  ha.breaker_W = (const float*)d_in[65]; ha.breaker_b = (const float*)d_in[66];
  ha.risk_W    = (const float*)d_in[67]; ha.risk_b    = (const float*)d_in[68];
  ha.value_W   = (const float*)d_in[69]; ha.value_b   = (const float*)d_in[70];
  ha.out = out;
  heads_kernel<<<1, 256, 0, stream>>>(ha);

  score_kernel<<<(20000 + 255) / 256, 256, 0, stream>>>(
      buf_h + nodeOff[1] * HG, (const float*)d_in[61], (const float*)d_in[62],
      out + 176, 20000);
  score_kernel<<<(10000 + 255) / 256, 256, 0, stream>>>(
      buf_h + nodeOff[2] * HG, (const float*)d_in[63], (const float*)d_in[64],
      out + 20176, 10000);
}
